// VQVAE_9388798509514
// MI455X (gfx1250) — matrix operand order repaired and checked by reading, not compiled, not run on hardware
//
#include <hip/hip_runtime.h>
#include <hip/hip_bf16.h>

typedef __attribute__((ext_vector_type(16))) _Float16 v16h;
typedef __attribute__((ext_vector_type(8)))  float    v8f;

// ---------------------------------------------------------------------------
// Weight repack: fp32 [Co,Ci,KH,KW] (or convT [Ci,Co,KH,KW] flipped) -> f16,
// FRAGMENT-ORDERED: dst[co][kchunk][slot] with the per-chunk k-group swap
// (0,1,2,3 -> 0,2,1,3; involution) baked in, Co padded to Copad (64-mult),
// K padded to Kpad (32-mult), zero-filled.  Each WMMA lane's A fragment is
// then a contiguous 32B run in global memory (L2-resident, no LDS staging).
// ---------------------------------------------------------------------------
__global__ void repack_w_kernel(const float* __restrict__ src, _Float16* __restrict__ dst,
                                int Co, int Ci, int KH, int KW, int tflip,
                                int Copad, int Kpad)
{
    long total = (long)Copad * Kpad;
    long i = (long)blockIdx.x * blockDim.x + threadIdx.x;
    if (i >= total) return;
    int co = (int)(i / Kpad);
    int t  = (int)(i % Kpad);
    int pj = t & 31;
    int gj = pj >> 3;
    int k  = (pj & 7) | ((((gj & 1) << 1) | (gj >> 1)) << 3);   // involution
    int kk = (t & ~31) + k;
    int Ktot = Ci * KH * KW;
    float v = 0.f;
    if (co < Co && kk < Ktot) {
        int KHW = KH * KW;
        int ci = kk / KHW;
        int r  = kk - ci * KHW;
        int kh = r / KW, kw = r - kh * KW;
        if (!tflip)
            v = src[(((long)co * Ci + ci) * KH + kh) * KW + kw];
        else
            v = src[(((long)ci * Co + co) * KH + (KH - 1 - kh)) * KW + (KW - 1 - kw)];
    }
    dst[i] = (_Float16)v;
}

// ---------------------------------------------------------------------------
// Implicit-GEMM conv (NCHW) with WMMA f16->f32.
//   Block tile: M=64 (Cout) x N=64 (pixels), K chunk 32.
//   8 waves; each wave: 1 A frag (direct contiguous global b128 from the
//   fragment-ordered weight pack) + 2 B frags -> 2 WMMAs per chunk.
//   B tile is double-buffered in LDS: one __syncthreads per chunk, im2col
//   gather of chunk k+1 overlaps the WMMAs of chunk k.
//   ldil = lhs dilation (1 = normal conv; s = transposed conv, with
//   stride=1 and pad = k-1-pad_orig).
// ---------------------------------------------------------------------------
__global__ __launch_bounds__(256)
void conv_wmma_kernel(const float* __restrict__ x, const _Float16* __restrict__ w16,
                      const float* __restrict__ bias, float* __restrict__ y,
                      int N, int Cin, int Hin, int Win,
                      int Cout, int Hout, int Wout,
                      int KH, int KW, int stride, int pad, int ldil, int Kpad)
{
    __shared__ alignas(32) _Float16 Bs[2][64 * 32];   // [n][k], double-buffered
    const int tid  = threadIdx.x;
    const int lane = tid & 31;
    const int wave = tid >> 5;
    const int lh   = lane & 15;
    const int sel  = lane >> 4;
    const int mo   = (wave & 3) * 16;        // 0,16,32,48
    const int nb   = (wave >> 2) * 32;       // 0,32
    const int no0  = nb, no1 = nb + 16;
    const int cobase  = blockIdx.y * 64;
    const long pixbase = (long)blockIdx.x * 64;
    const int  Ktot = Cin * KH * KW;
    const long Npix = (long)N * Hout * Wout;
    const int  KHW  = KH * KW;
    const _Float16* arow = w16 + (long)(cobase + mo + lh) * Kpad + sel * 16;

    auto stageB = [&](int kc, _Float16* Bbuf) {
        #pragma unroll
        for (int j = 0; j < 8; ++j) {
            int idx = tid + 256 * j;           // 0..2047
            int nl = idx & 63, k = idx >> 6;
            int kk = kc + k;
            long p = pixbase + nl;
            float v = 0.f;
            if (kk < Ktot && p < Npix) {
                int ci = kk / KHW;
                int r  = kk - ci * KHW;
                int kh = r / KW, kw = r - kh * KW;
                int ow = (int)(p % Wout);
                long t = p / Wout;
                int oh = (int)(t % Hout);
                int n  = (int)(t / Hout);
                int th = oh * stride - pad + kh;
                int tw = ow * stride - pad + kw;
                if (th >= 0 && tw >= 0 && (th % ldil) == 0 && (tw % ldil) == 0) {
                    int ih = th / ldil, iw = tw / ldil;
                    if (ih < Hin && iw < Win)
                        v = x[(((long)n * Cin + ci) * Hin + ih) * Win + iw];
                }
            }
            Bbuf[nl * 32 + ((k & 7) | (((((k >> 3) & 1) << 1) | (k >> 4)) << 3))] = (_Float16)v;
        }
    };

    v8f c0 = {}, c1 = {};
    stageB(0, Bs[0]);
    int cur = 0;
    for (int kc = 0; kc < Kpad; kc += 32) {
        __syncthreads();                       // staged chunk kc now visible
        if (kc + 32 < Kpad) {
            __builtin_prefetch(arow + kc + 32, 0, 1);   // global_prefetch next A
            stageB(kc + 32, Bs[cur ^ 1]);      // overlap gather with math
        }
        // A fragment: contiguous 32B global chunk (L2-resident weights)
        v16h a = *reinterpret_cast<const v16h*>(arow + kc);
        const v16h* Bv = reinterpret_cast<const v16h*>(Bs[cur]);
        v16h b0 = Bv[(no0 + lh) * 2 + sel];
        v16h b1 = Bv[(no1 + lh) * 2 + sel];
        c0 = __builtin_amdgcn_wmma_f32_16x16x32_f16(false, a, false, b0, (short)0, c0,
                                                    false, false);
        c1 = __builtin_amdgcn_wmma_f32_16x16x32_f16(false, a, false, b1, (short)0, c1,
                                                    false, false);
        cur ^= 1;
    }
    // ---- epilogue: D layout -> NCHW fp32 + bias ----
    #pragma unroll
    for (int r = 0; r < 8; ++r) {
        int co = cobase + mo + r + 8 * sel;
        if (co < Cout) {
            float bv = bias[co];
            long p0 = pixbase + no0 + lh;
            if (p0 < Npix) {
                int ow = (int)(p0 % Wout);
                long t = p0 / Wout;
                int oh = (int)(t % Hout);
                int n  = (int)(t / Hout);
                y[(((long)n * Cout + co) * Hout + oh) * Wout + ow] = c0[r] + bv;
            }
            long p1 = pixbase + no1 + lh;
            if (p1 < Npix) {
                int ow = (int)(p1 % Wout);
                long t = p1 / Wout;
                int oh = (int)(t % Hout);
                int n  = (int)(t / Hout);
                y[(((long)n * Cout + co) * Hout + oh) * Wout + ow] = c1[r] + bv;
            }
        }
    }
}

// ---------------------------------------------------------------------------
// BatchNorm training-mode stats: one block per channel.
// stat[c] = mean, stat[C+c] = rsqrt(biased_var + eps)
// ---------------------------------------------------------------------------
__global__ __launch_bounds__(256)
void bn_stats_kernel(const float* __restrict__ x, float* __restrict__ stat,
                     int C, int HW, int N)
{
    __shared__ float ss[256], ss2[256];
    int c = blockIdx.x;
    long cnt = (long)N * HW;
    float s = 0.f, s2 = 0.f;
    const float4* x4 = (const float4*)x;
    int HW4 = HW >> 2;
    long cnt4 = (long)N * HW4;
    for (long i = threadIdx.x; i < cnt4; i += 256) {
        long n = i / HW4, r = i % HW4;
        float4 v = x4[(n * C + c) * HW4 + r];
        s  += v.x + v.y + v.z + v.w;
        s2 += v.x * v.x + v.y * v.y + v.z * v.z + v.w * v.w;
    }
    ss[threadIdx.x] = s; ss2[threadIdx.x] = s2;
    __syncthreads();
    for (int o = 128; o > 0; o >>= 1) {
        if (threadIdx.x < o) {
            ss[threadIdx.x]  += ss[threadIdx.x + o];
            ss2[threadIdx.x] += ss2[threadIdx.x + o];
        }
        __syncthreads();
    }
    if (threadIdx.x == 0) {
        float m   = ss[0] / (float)cnt;
        float var = ss2[0] / (float)cnt - m * m;
        stat[c]     = m;
        stat[C + c] = rsqrtf(var + 1e-5f);
    }
}

// Fused BN apply + optional ReLU + optional residual add (float4 vectorized;
// HW is always a multiple of 4 so the quad shares one channel).
__global__ void bn_apply_kernel(const float* __restrict__ x, const float* __restrict__ stat,
                                const float* __restrict__ g, const float* __restrict__ b,
                                const float* __restrict__ res, float* __restrict__ out,
                                int C, int HW, long total4, int relu)
{
    long i = (long)blockIdx.x * blockDim.x + threadIdx.x;
    if (i >= total4) return;
    int c = (int)(((i * 4) / HW) % C);
    float sc = stat[C + c] * g[c];
    float mb = b[c] - stat[c] * sc;
    float4 v = ((const float4*)x)[i];
    float4 o;
    o.x = v.x * sc + mb; o.y = v.y * sc + mb; o.z = v.z * sc + mb; o.w = v.w * sc + mb;
    if (relu) {
        o.x = o.x > 0.f ? o.x : 0.f; o.y = o.y > 0.f ? o.y : 0.f;
        o.z = o.z > 0.f ? o.z : 0.f; o.w = o.w > 0.f ? o.w : 0.f;
    }
    if (res) {
        float4 rv = ((const float4*)res)[i];
        o.x += rv.x; o.y += rv.y; o.z += rv.z; o.w += rv.w;
    }
    ((float4*)out)[i] = o;
}

// ---------------------------------------------------------------------------
// VQ: fragment-major pre-pack of queries and codebook (f32 -> f16) so the
// argmin wave issues only contiguous 32B vector loads.
// ---------------------------------------------------------------------------
__global__ void qpack_kernel(const float* __restrict__ enc, _Float16* __restrict__ qp, long total)
{
    long idx = (long)blockIdx.x * blockDim.x + threadIdx.x;
    if (idx >= total) return;
    int  j    = (int)(idx & 31);
    int  lane = (int)((idx >> 5) & 31);
    long blk  = idx >> 10;
    int lh = lane & 15, sel = lane >> 4;
    int jj = j & 15;
    int k  = (j >= 16 ? 32 : 0) + jj + 8 * (sel + (jj >= 8 ? 1 : 0));
    qp[idx] = (_Float16)enc[(blk * 16 + lh) * 64 + k];
}

__global__ void epack_kernel(const float* __restrict__ emb, _Float16* __restrict__ ep)
{
    int idx = blockIdx.x * blockDim.x + threadIdx.x;   // 32768 total
    if (idx >= 32768) return;
    int j    = idx & 31;
    int lane = (idx >> 5) & 31;
    int ct   = idx >> 10;
    int lh = lane & 15, sel = lane >> 4;
    int code = ct * 16 + lh;
    int jj = j & 15;
    int kb = (j >= 16 ? 32 : 0) + jj + 8 * (sel + (jj >= 8 ? 1 : 0));
    ep[idx] = (_Float16)emb[code * 64 + kb];
}

__global__ void emb_norm_kernel(const float* __restrict__ emb, float* __restrict__ enorm)
{
    int k = blockIdx.x * blockDim.x + threadIdx.x;
    if (k < 512) {
        float s = 0.f;
        for (int d = 0; d < 64; ++d) { float v = emb[k * 64 + d]; s += v * v; }
        enorm[k] = s;
    }
}

__global__ void init_loss_kernel(float* lossacc) { if (threadIdx.x == 0) lossacc[0] = 0.f; }

// One wave per 16 query rows: per 16-code tile, two k32 WMMAs accumulate
// Q[16x64] x E^T[64x16]; argmin of ||e||^2 - 2 q.e over 512 codes,
// shuffle-reduced across the 16 lanes holding different codes.
__global__ __launch_bounds__(32)
void vq_argmin_kernel(const _Float16* __restrict__ qp, const _Float16* __restrict__ ep,
                      const float* __restrict__ enorm, int* __restrict__ closest)
{
    const int lane = threadIdx.x & 31;
    const int lh = lane & 15, sel = lane >> 4;
    const long rowbase = (long)blockIdx.x * 16;
    const v16h* qv = reinterpret_cast<const v16h*>(qp) + ((long)blockIdx.x * 32 + lane) * 2;
    v16h a0 = qv[0];
    v16h a1 = qv[1];
    float best[8]; int besti[8];
    #pragma unroll
    for (int r = 0; r < 8; ++r) { best[r] = 3.4e38f; besti[r] = 0; }
    for (int ct = 0; ct < 32; ++ct) {
        const v16h* ev = reinterpret_cast<const v16h*>(ep) + ((long)ct * 32 + lane) * 2;
        v16h b0 = ev[0];
        v16h b1 = ev[1];
        v8f g = {};
        g = __builtin_amdgcn_wmma_f32_16x16x32_f16(false, a0, false, b0, (short)0, g, false, false);
        g = __builtin_amdgcn_wmma_f32_16x16x32_f16(false, a1, false, b1, (short)0, g, false, false);
        int code = ct * 16 + lh;
        float en = enorm[code];
        #pragma unroll
        for (int r = 0; r < 8; ++r) {
            float d = en - 2.f * g[r];
            if (d < best[r] || (d == best[r] && code < besti[r])) { best[r] = d; besti[r] = code; }
        }
    }
    #pragma unroll
    for (int m = 1; m < 16; m <<= 1) {
        #pragma unroll
        for (int r = 0; r < 8; ++r) {
            float ov = __shfl_xor(best[r], m, 32);
            int   oi = __shfl_xor(besti[r], m, 32);
            if (ov < best[r] || (ov == best[r] && oi < besti[r])) { best[r] = ov; besti[r] = oi; }
        }
    }
    if (lh == 0) {
        #pragma unroll
        for (int r = 0; r < 8; ++r)
            closest[rowbase + r + 8 * sel] = besti[r];
    }
}

// quant = emb[closest]; accumulate sum((quant - qi)^2) for the loss.
__global__ __launch_bounds__(256)
void vq_gather_loss_kernel(const float* __restrict__ emb, const int* __restrict__ closest,
                           const float* __restrict__ enc, float* __restrict__ quant,
                           float* __restrict__ lossacc, long total)
{
    __shared__ float red[256];
    long i = (long)blockIdx.x * 256 + threadIdx.x;
    float p = 0.f;
    if (i < total) {
        long row = i >> 6; int c = (int)(i & 63);
        float q = emb[(long)closest[row] * 64 + c];
        quant[i] = q;
        float d = q - enc[i];
        p = d * d;
    }
    red[threadIdx.x] = p;
    __syncthreads();
    for (int o = 128; o > 0; o >>= 1) {
        if (threadIdx.x < o) red[threadIdx.x] += red[threadIdx.x + o];
        __syncthreads();
    }
    if (threadIdx.x == 0) atomicAdd(lossacc, red[0]);
}

// closest -> d_out[1572864..] as floats; forward value of codebook loss ==
// commitment loss => quantize_loss = 1.25 * mse -> d_out[1605632].
__global__ void vq_finalize_kernel(const int* __restrict__ closest,
                                   const float* __restrict__ lossacc,
                                   float* __restrict__ dout)
{
    int i = blockIdx.x * blockDim.x + threadIdx.x;
    if (i < 32768) dout[1572864 + i] = (float)closest[i];
    if (i == 0)    dout[1605632] = 1.25f * lossacc[0] / 2097152.0f;
}

// ---------------------------------------------------------------------------
// Host-side helpers (launch only; graph-capture safe)
// ---------------------------------------------------------------------------
static void launch_conv_stage(hipStream_t s, const float* src_w, _Float16* wpack,
                              const float* x, const float* bias, float* y,
                              int N, int Cin, int Hin, int Win,
                              int Cout, int Hout, int Wout, int KH, int KW,
                              int stride, int pad, int ldil, int tflip)
{
    int Ktot  = Cin * KH * KW;
    int Kpad  = (Ktot + 31) & ~31;
    int Copad = (Cout + 63) & ~63;
    long wtotal = (long)Copad * Kpad;
    repack_w_kernel<<<(unsigned)((wtotal + 255) / 256), 256, 0, s>>>(
        src_w, wpack, Cout, Cin, KH, KW, tflip, Copad, Kpad);
    long Npix = (long)N * Hout * Wout;
    dim3 grid((unsigned)((Npix + 63) / 64), (unsigned)(Copad / 64));
    conv_wmma_kernel<<<grid, 256, 0, s>>>(x, wpack, bias, y, N, Cin, Hin, Win,
                                          Cout, Hout, Wout, KH, KW, stride, pad, ldil, Kpad);
}

static void launch_bn(hipStream_t s, const float* x, float* stat,
                      const float* g, const float* b, const float* res, float* out,
                      int C, int HW, int N, int relu)
{
    bn_stats_kernel<<<C, 256, 0, s>>>(x, stat, C, HW, N);
    long total4 = ((long)N * C * HW) >> 2;
    bn_apply_kernel<<<(unsigned)((total4 + 255) / 256), 256, 0, s>>>(x, stat, g, b, res, out,
                                                                     C, HW, total4, relu);
}

extern "C" void kernel_launch(void* const* d_in, const int* in_sizes, int n_in,
                              void* d_out, int out_size, void* d_ws, size_t ws_size,
                              hipStream_t stream)
{
    const float* x      = (const float*)d_in[0];
    const float* ec1_w  = (const float*)d_in[1];  const float* ec1_b  = (const float*)d_in[2];
    const float* ebn1_g = (const float*)d_in[3];  const float* ebn1_b = (const float*)d_in[4];
    const float* ec2_w  = (const float*)d_in[5];  const float* ec2_b  = (const float*)d_in[6];
    const float* ebn2_g = (const float*)d_in[7];  const float* ebn2_b = (const float*)d_in[8];
    const float* er1_w  = (const float*)d_in[9];  const float* er1_b  = (const float*)d_in[10];
    const float* ebnr1_g= (const float*)d_in[11]; const float* ebnr1_b= (const float*)d_in[12];
    const float* er2_w  = (const float*)d_in[13]; const float* er2_b  = (const float*)d_in[14];
    const float* ebnr2_g= (const float*)d_in[15]; const float* ebnr2_b= (const float*)d_in[16];
    const float* eproj_w= (const float*)d_in[17]; const float* eproj_b= (const float*)d_in[18];
    const float* emb    = (const float*)d_in[19];
    const float* dproj_w= (const float*)d_in[20]; const float* dproj_b= (const float*)d_in[21];
    const float* dr1_w  = (const float*)d_in[22]; const float* dr1_b  = (const float*)d_in[23];
    const float* dbn1_g = (const float*)d_in[24]; const float* dbn1_b = (const float*)d_in[25];
    const float* dr2_w  = (const float*)d_in[26]; const float* dr2_b  = (const float*)d_in[27];
    const float* dbn2_g = (const float*)d_in[28]; const float* dbn2_b = (const float*)d_in[29];
    const float* dt1_w  = (const float*)d_in[30]; const float* dt1_b  = (const float*)d_in[31];
    const float* dbn3_g = (const float*)d_in[32]; const float* dbn3_b = (const float*)d_in[33];
    const float* dt2_w  = (const float*)d_in[34]; const float* dt2_b  = (const float*)d_in[35];
    float* out = (float*)d_out;

    // ---- workspace layout (floats) ----
    float* W = (float*)d_ws;
    const size_t T1 = 33554432ul;  // 8*256*128*128
    const size_t T2 = 4194304ul;   // 8*128*64*64
    const size_t TE = 2097152ul;   // 8*64*64*64
    float* A0  = W;                // stage-1 conv out / dt1 out
    float* A1  = W + T1;           // stage-1 bn out / dt1 bn out
    float* Bb0 = W + 2 * T1;
    float* Bb1 = Bb0 + T2;
    float* Bb2 = Bb1 + T2;         // conv scratch for residual branches
    float* Bb3 = Bb2 + T2;
    float* Bb4 = Bb3 + T2;
    float* Bb5 = Bb4 + T2;
    float* Bb6 = Bb5 + T2;
    float* Bb7 = Bb6 + T2;
    float* enc   = Bb7 + T2;       // [8,64,64,64]
    float* quant = enc + TE;
    _Float16* qp = (_Float16*)(quant + TE);           // TE halves (frag-major queries)
    float* after_qp = quant + TE + TE / 2;
    _Float16* ep = (_Float16*)after_qp;               // 32768 halves (frag-major codes)
    float* enorm = after_qp + 16384;
    int*   closest = (int*)(enorm + 512);             // 32768 ints
    float* bnstat  = (float*)(closest + 32768);       // 512 floats (mean|invstd)
    _Float16* wpack = (_Float16*)(bnstat + 512);      // up to 524288 halves (frag-ordered)
    float* lossacc = bnstat + 512 + 262144;

    // ================= encoder =================
    launch_conv_stage(stream, ec1_w, wpack, x, ec1_b, A0,
                      8, 3, 256, 256, 256, 128, 128, 4, 4, 2, 1, 1, 0);
    launch_bn(stream, A0, bnstat, ebn1_g, ebn1_b, nullptr, A1, 256, 128 * 128, 8, 1);

    launch_conv_stage(stream, ec2_w, wpack, A1, ec2_b, Bb0,
                      8, 256, 128, 128, 128, 64, 64, 4, 4, 2, 1, 1, 0);
    launch_bn(stream, Bb0, bnstat, ebn2_g, ebn2_b, nullptr, Bb1, 128, 64 * 64, 8, 1);

    launch_conv_stage(stream, er1_w, wpack, Bb1, er1_b, Bb2,
                      8, 128, 64, 64, 128, 64, 64, 3, 3, 1, 1, 1, 0);
    launch_bn(stream, Bb2, bnstat, ebnr1_g, ebnr1_b, Bb1, Bb3, 128, 64 * 64, 8, 1);

    launch_conv_stage(stream, er2_w, wpack, Bb3, er2_b, Bb2,
                      8, 128, 64, 64, 128, 64, 64, 1, 1, 1, 0, 1, 0);
    launch_bn(stream, Bb2, bnstat, ebnr2_g, ebnr2_b, Bb3, Bb4, 128, 64 * 64, 8, 1);

    launch_conv_stage(stream, eproj_w, wpack, Bb4, eproj_b, enc,
                      8, 128, 64, 64, 64, 64, 64, 1, 1, 1, 0, 1, 0);

    // ================= vector quantization =================
    qpack_kernel<<<(unsigned)((TE + 255) / 256), 256, 0, stream>>>(enc, qp, (long)TE);
    epack_kernel<<<128, 256, 0, stream>>>(emb, ep);
    emb_norm_kernel<<<2, 256, 0, stream>>>(emb, enorm);
    init_loss_kernel<<<1, 32, 0, stream>>>(lossacc);
    vq_argmin_kernel<<<2048, 32, 0, stream>>>(qp, ep, enorm, closest);
    vq_gather_loss_kernel<<<(unsigned)((TE + 255) / 256), 256, 0, stream>>>(
        emb, closest, enc, quant, lossacc, (long)TE);
    vq_finalize_kernel<<<128, 256, 0, stream>>>(closest, lossacc, out);

    // ================= decoder =================
    launch_conv_stage(stream, dproj_w, wpack, quant, dproj_b, Bb5,
                      8, 64, 64, 64, 128, 64, 64, 1, 1, 1, 0, 1, 0);

    launch_conv_stage(stream, dr1_w, wpack, Bb5, dr1_b, Bb2,
                      8, 128, 64, 64, 128, 64, 64, 3, 3, 1, 1, 1, 0);
    launch_bn(stream, Bb2, bnstat, dbn1_g, dbn1_b, Bb5, Bb6, 128, 64 * 64, 8, 1);

    launch_conv_stage(stream, dr2_w, wpack, Bb6, dr2_b, Bb2,
                      8, 128, 64, 64, 128, 64, 64, 3, 3, 1, 1, 1, 0);
    launch_bn(stream, Bb2, bnstat, dbn2_g, dbn2_b, Bb6, Bb7, 128, 64 * 64, 8, 1);

    // dt1: ConvT(128->256, k4, s2, p1)  ==  conv(stride 1, pad 2, ldil 2)
    launch_conv_stage(stream, dt1_w, wpack, Bb7, dt1_b, A0,
                      8, 128, 64, 64, 256, 128, 128, 4, 4, 1, 2, 2, 1);
    launch_bn(stream, A0, bnstat, dbn3_g, dbn3_b, nullptr, A1, 256, 128 * 128, 8, 1);

    // dt2: ConvT(256->3, k4, s2, p1) -> final output directly into d_out
    launch_conv_stage(stream, dt2_w, wpack, A1, dt2_b, out,
                      8, 256, 128, 128, 3, 256, 256, 4, 4, 1, 2, 2, 1);
}
